// GCNNew_78984448573632
// MI455X (gfx1250) — matrix operand order repaired and checked by reading, not compiled, not run on hardware
//
#include <hip/hip_runtime.h>
#include <math.h>

// ---------------- problem constants ----------------
#define N_NODES   100000
#define N_EDGES   3200000
#define EMBED_DIM 128
#define ATTR_DIM  8
#define K_RAW     (EMBED_DIM + ATTR_DIM)   // 136
#define K_PAD     160                      // 5 WMMA K-steps of 32
#define LDS_STRIDE 168                     // padded LDS row stride (bank-conflict free)
#define M_TILES   (N_NODES / 16)           // 6250 exactly

typedef __attribute__((ext_vector_type(16))) __bf16 v16bf;
typedef __attribute__((ext_vector_type(8)))  __bf16 v8bf;
typedef __attribute__((ext_vector_type(8)))  float  v8f;
typedef __attribute__((ext_vector_type(4)))  int    v4i;

#define __AS(n) __attribute__((address_space(n)))

#if __has_builtin(__builtin_amdgcn_global_load_async_to_lds_b128) && \
    __has_builtin(__builtin_amdgcn_s_wait_asynccnt)
#define HAVE_ASYNC_LDS 1
#else
#define HAVE_ASYNC_LDS 0
#endif

// ---------------- helpers ----------------
__device__ __forceinline__ void atomic_add_f32(float* p, float v) {
    unsafeAtomicAdd(p, v);   // hardware global_atomic_add_f32 (no CAS loop)
}

__device__ __forceinline__ v16bf join16(v8bf lo, v8bf hi) {
    return __builtin_shufflevector(lo, hi,
             0,1,2,3,4,5,6,7,8,9,10,11,12,13,14,15);
}

// ---------------- pack concat(X, attr) -> bf16 [N, K_PAD] ----------------
__global__ void pack_x_kernel(const float* __restrict__ X,
                              const float* __restrict__ attr,
                              __bf16* __restrict__ Xc) {
    int idx = blockIdx.x * blockDim.x + threadIdx.x;
    if (idx >= N_NODES * K_PAD) return;
    int i = idx / K_PAD;
    int k = idx - i * K_PAD;
    float v;
    if (k < EMBED_DIM)       v = X[(size_t)i * EMBED_DIM + k];
    else if (k < K_RAW)      v = attr[(size_t)i * ATTR_DIM + (k - EMBED_DIM)];
    else                     v = 0.0f;
    Xc[idx] = (__bf16)v;
}

// ---------------- pack W [136,128] -> W^T bf16 [128, K_PAD] ----------------
__global__ void pack_w_kernel(const float* __restrict__ W,
                              __bf16* __restrict__ Wt) {
    int idx = blockIdx.x * blockDim.x + threadIdx.x;
    if (idx >= EMBED_DIM * K_PAD) return;
    int n = idx / K_PAD;
    int k = idx - n * K_PAD;
    float v = (k < K_RAW) ? W[(size_t)k * EMBED_DIM + n] : 0.0f;
    Wt[idx] = (__bf16)v;
}

// ---------------- dense GEMM: H[N,128] = Xc[N,160] @ Wt^T via WMMA bf16 ----
// W^T staged once per block into LDS (async copy); one wave per 16x128 slab.
__global__ void __launch_bounds__(256) gemm_wmma_kernel(
        const __bf16* __restrict__ Xc,
        const __bf16* __restrict__ Wt,
        float* __restrict__ H) {
    __shared__ __bf16 sW[EMBED_DIM * LDS_STRIDE];   // 128 rows x 168 bf16 = 43008 B

    // ---- stage W^T (128x160 used region) into LDS, 16B chunks ----
    // 128*160/8 = 2560 chunks, 256 threads -> 10 chunks each
    {
        int t = threadIdx.x;
#pragma unroll
        for (int it = 0; it < 10; ++it) {
            int c  = it * 256 + t;          // chunk id
            int n  = c / 20;                // row (20 chunks of 8 per row)
            int kc = (c - n * 20) * 8;      // element offset within row
            const __bf16* src = Wt + (size_t)n * K_PAD + kc;
            __bf16*       dst = &sW[n * LDS_STRIDE + kc];
#if HAVE_ASYNC_LDS
            __builtin_amdgcn_global_load_async_to_lds_b128(
                (__AS(1) v4i*)const_cast<__bf16*>(src),
                (__AS(3) v4i*)dst, 0, 0);
#else
            *(v8bf*)dst = *(const v8bf*)src;
#endif
        }
#if HAVE_ASYNC_LDS
        __builtin_amdgcn_s_wait_asynccnt(0);
#endif
        __syncthreads();
    }

    int wid  = blockIdx.x * (blockDim.x >> 5) + (threadIdx.x >> 5);
    if (wid >= M_TILES) return;
    int lane = threadIdx.x & 31;
    int mr   = lane & 15;        // M (for A) / N (for B) index within tile
    int hi   = lane >> 4;        // lane-group select
    int m0   = wid * 16;

    // A fragment: lane<16 -> K {0..7,16..23}; lane>=16 -> K {8..15,24..31}
    const __bf16* arow = Xc + (size_t)(m0 + mr) * K_PAD + hi * 8;

    v8f acc[8] = {};
#pragma unroll
    for (int ks = 0; ks < K_PAD / 32; ++ks) {
        const __bf16* ap = arow + ks * 32;
        v16bf a = join16(*(const v8bf*)(ap), *(const v8bf*)(ap + 16));
#pragma unroll
        for (int n = 0; n < 8; ++n) {
            // B fragment from LDS: lane<16 -> K {0..7,16..23}; lane>=16 -> K {8..15,24..31}
            const __bf16* bp = &sW[(n * 16 + mr) * LDS_STRIDE + ks * 32 + hi * 8];
            v16bf b = join16(*(const v8bf*)(bp), *(const v8bf*)(bp + 16));
            acc[n] = __builtin_amdgcn_wmma_f32_16x16x32_bf16(
                         false, a, false, b, (short)0, acc[n], false, false);
        }
    }
    // C/D layout: lane<16 -> N=lane, M=j; lane>=16 -> N=lane-16, M=j+8
#pragma unroll
    for (int n = 0; n < 8; ++n) {
#pragma unroll
        for (int j = 0; j < 8; ++j) {
            H[(size_t)(m0 + j + 8 * hi) * EMBED_DIM + n * 16 + mr] = acc[n][j];
        }
    }
}

// ---------------- COO SpMM: Y[r] += vals[e] * H[c]  (wave per edge) -------
__global__ void spmm_atomic_kernel(const int* __restrict__ rows,
                                   const int* __restrict__ cols,
                                   const float* __restrict__ vals,
                                   const float* __restrict__ H,
                                   float* __restrict__ Y) {
    int wid  = blockIdx.x * (blockDim.x >> 5) + (threadIdx.x >> 5);
    int lane = threadIdx.x & 31;
    int nw   = gridDim.x * (blockDim.x >> 5);
    for (int e = wid; e < N_EDGES; e += nw) {
        int   r = rows[e];
        int   c = cols[e];
        float v = vals[e];
        float4 x = ((const float4*)(H + (size_t)c * EMBED_DIM))[lane];
        float* dst = Y + (size_t)r * EMBED_DIM + lane * 4;
        atomic_add_f32(dst + 0, v * x.x);
        atomic_add_f32(dst + 1, v * x.y);
        atomic_add_f32(dst + 2, v * x.z);
        atomic_add_f32(dst + 3, v * x.w);
        int e2 = e + nw;
        if (e2 < N_EDGES) {
            int c2 = cols[e2];
            __builtin_prefetch(H + (size_t)c2 * EMBED_DIM + lane * 4, 0, 0);
        }
    }
}

// ---------------- tanh + repack into bf16 input for next layer ------------
__global__ void act_pack_kernel(const float* __restrict__ Y,
                                __bf16* __restrict__ Xc) {
    int idx = blockIdx.x * blockDim.x + threadIdx.x;
    if (idx >= N_NODES * EMBED_DIM) return;
    int i = idx >> 7;          // / 128
    int d = idx & 127;
    Xc[(size_t)i * K_PAD + d] = (__bf16)tanhf(Y[idx]);
}

// ---------------- tanh + L2 row-normalize (wave per node) -----------------
__global__ void finalize_kernel(const float* __restrict__ Y,
                                float* __restrict__ out) {
    int wid  = blockIdx.x * (blockDim.x >> 5) + (threadIdx.x >> 5);
    if (wid >= N_NODES) return;
    int lane = threadIdx.x & 31;
    float4 y = ((const float4*)(Y + (size_t)wid * EMBED_DIM))[lane];
    float t0 = tanhf(y.x), t1 = tanhf(y.y), t2 = tanhf(y.z), t3 = tanhf(y.w);
    float s = t0 * t0 + t1 * t1 + t2 * t2 + t3 * t3;
#pragma unroll
    for (int off = 16; off > 0; off >>= 1)
        s += __shfl_xor(s, off, 32);
    float inv = 1.0f / sqrtf(fmaxf(s, 1e-12f));
    float4 o = make_float4(t0 * inv, t1 * inv, t2 * inv, t3 * inv);
    ((float4*)(out + (size_t)wid * EMBED_DIM))[lane] = o;
}

// ---------------- host-side launcher --------------------------------------
extern "C" void kernel_launch(void* const* d_in, const int* in_sizes, int n_in,
                              void* d_out, int out_size, void* d_ws, size_t ws_size,
                              hipStream_t stream) {
    const int*   rows  = (const int*)  d_in[0];
    const int*   cols  = (const int*)  d_in[1];
    const float* vals  = (const float*)d_in[2];
    const float* attr  = (const float*)d_in[3];
    const float* X     = (const float*)d_in[4];
    const float* W0    = (const float*)d_in[5];
    const float* W1    = (const float*)d_in[6];
    float*       out   = (float*)d_out;

    // workspace carve-up (all offsets 256B aligned)
    char* ws = (char*)d_ws;
    const size_t xcBytes = (size_t)N_NODES * K_PAD * sizeof(__bf16);     // 32,000,000
    const size_t wtBytes = (size_t)EMBED_DIM * K_PAD * sizeof(__bf16);   // 40,960
    const size_t hBytes  = (size_t)N_NODES * EMBED_DIM * sizeof(float);  // 51,200,000
    __bf16* Xc  = (__bf16*)(ws);
    __bf16* W0t = (__bf16*)(ws + xcBytes);
    __bf16* W1t = (__bf16*)(ws + xcBytes + wtBytes);
    float*  H   = (float*) (ws + xcBytes + 2 * wtBytes);
    float*  Y   = (float*) (ws + xcBytes + 2 * wtBytes + hBytes);

    const int TB = 256;
    const int gemmBlocks  = (M_TILES + 7) / 8;                 // 8 waves/block
    const int spmmBlocks  = 4096;                              // 32768 waves, grid-stride
    const int packXBlocks = (N_NODES * K_PAD + TB - 1) / TB;
    const int packWBlocks = (EMBED_DIM * K_PAD + TB - 1) / TB;
    const int actBlocks   = (N_NODES * EMBED_DIM + TB - 1) / TB;
    const int finBlocks   = N_NODES / 8;                       // 8 waves/block

    // pack inputs
    pack_x_kernel<<<packXBlocks, TB, 0, stream>>>(X, attr, Xc);
    pack_w_kernel<<<packWBlocks, TB, 0, stream>>>(W0, W0t);
    pack_w_kernel<<<packWBlocks, TB, 0, stream>>>(W1, W1t);

    // ---- layer 0 ----
    gemm_wmma_kernel<<<gemmBlocks, TB, 0, stream>>>(Xc, W0t, H);
    (void)hipMemsetAsync(Y, 0, hBytes, stream);
    spmm_atomic_kernel<<<spmmBlocks, TB, 0, stream>>>(rows, cols, vals, H, Y);
    act_pack_kernel<<<actBlocks, TB, 0, stream>>>(Y, Xc);

    // ---- layer 1 ----
    gemm_wmma_kernel<<<gemmBlocks, TB, 0, stream>>>(Xc, W1t, H);
    (void)hipMemsetAsync(Y, 0, hBytes, stream);
    spmm_atomic_kernel<<<spmmBlocks, TB, 0, stream>>>(rows, cols, vals, H, Y);

    // ---- tanh + L2 normalize ----
    finalize_kernel<<<finBlocks, TB, 0, stream>>>(Y, out);
}
